// POLAAgent_46179488367179
// MI455X (gfx1250) — matrix operand order repaired and checked by reading, not compiled, not run on hardware
//
#include <hip/hip_runtime.h>
#include <hip/hip_bf16.h>

typedef __attribute__((ext_vector_type(16))) _Float16 v16h;
typedef __attribute__((ext_vector_type(8)))  float    v8f;

#define WMMA_F32_F16(a, b, c) \
    __builtin_amdgcn_wmma_f32_16x16x32_f16(false, (a), false, (b), (short)0, (c), false, false)

// A-register layout for a 16x32 f16 tile (CDNA5 ISA §7.12.2):
//   lane = m + 16*((k>>3)&1)   (m = row 0..15, k = 0..31)
//   elem = (k&7) + 8*(k>>4)    (element within the v16h)
__device__ __forceinline__ int a_lane(int m, int k) { return m + 16 * ((k >> 3) & 1); }
__device__ __forceinline__ int a_elem(int k)        { return (k & 7) + 8 * (k >> 4); }

// B-register layout for a 32x16 f16 tile:
//   lane = n + 16*hh, element e corresponds to k = 8*hh + (e&7) + 16*(e>>3)  (same as A)

// ---------------------------------------------------------------------------
// Kernel 0: weight precondition. Packs every WMMA B tile into per-lane
// register order so the GEMM kernels load each tile as one 32B contiguous
// global_load_b128 pair. Also pads W1 (K 36->64), permutes/pads Wend.
// ---------------------------------------------------------------------------
__global__ void k_prep(const float* __restrict__ W1, const float* __restrict__ Wir,
                       const float* __restrict__ Wiz, const float* __restrict__ Win,
                       const float* __restrict__ Wend, const float* __restrict__ bend,
                       _Float16* W1p, _Float16* Wirp, _Float16* Wizp, _Float16* Winp,
                       _Float16* Wendp, float* bendp) {
    const int PERM[4] = {1, 0, 3, 2};
    int tid    = threadIdx.x + blockIdx.x * blockDim.x;
    int stride = blockDim.x * gridDim.x;
    // 64x64 matrices: 8 tiles (kt 0..1, nt 0..3) x 32 lanes x 16 elems = 4096
    for (int p = tid; p < 4096; p += stride) {
        int e    = p & 15;
        int lane = (p >> 4) & 31;
        int t    = p >> 9;            // tile index = kt*4 + nt
        int kt   = t >> 2, nt = t & 3;
        int n    = lane & 15, hh = lane >> 4;
        int k    = kt * 32 + hh * 8 + (e & 7) + 16 * (e >> 3);
        int col  = nt * 16 + n;
        W1p[p]  = (_Float16)((k < 36) ? W1[k * 64 + col] : 0.0f);
        Wirp[p] = (_Float16)Wir[k * 64 + col];
        Wizp[p] = (_Float16)Wiz[k * 64 + col];
        Winp[p] = (_Float16)Win[k * 64 + col];
    }
    // Wend: [64,4] -> permuted/padded [64,16], 2 tiles (kt 0..1) x 32 x 16 = 1024
    for (int p = tid; p < 1024; p += stride) {
        int e    = p & 15;
        int lane = (p >> 4) & 31;
        int kt   = p >> 9;
        int n    = lane & 15, hh = lane >> 4;
        int k    = kt * 32 + hh * 8 + (e & 7) + 16 * (e >> 3);
        Wendp[p] = (_Float16)((n < 4) ? Wend[k * 4 + PERM[n]] : 0.0f);
    }
    if (tid < 16) bendp[tid] = (tid < 4) ? bend[PERM[tid]] : 0.0f;
}

// ---------------------------------------------------------------------------
// Kernel 1: hs = relu(x@W1+b1), then xr/xz/xn = hs@W{ir,iz,in} + b (WMMA).
// One wave per 16-row tile; A tiles staged in LDS directly in register layout
// (scatter 16-bit stores, contiguous 32B vector reads). B tiles are packed.
// ---------------------------------------------------------------------------
__global__ __launch_bounds__(256) void k_gates(
        const float* __restrict__ obs,                 // [T, 36]
        const float* __restrict__ b1,
        const float* __restrict__ bir, const float* __restrict__ biz,
        const float* __restrict__ bin,
        const v16h* __restrict__ W1p, const v16h* __restrict__ Wirp,
        const v16h* __restrict__ Wizp, const v16h* __restrict__ Winp,
        float* __restrict__ xr, float* __restrict__ xz, float* __restrict__ xn,
        int Tn) {
    __shared__ v16h sxA[8][2][32];   // [wave][ktile][lane] A-register image of x
    __shared__ v16h shA[8][2][32];   // [wave][ktile][lane] A-register image of hs
    int lane = threadIdx.x & 31;
    int wave = threadIdx.x >> 5;
    int t0   = (blockIdx.x * 8 + wave) * 16;
    int n    = lane & 15;
    int hh   = lane >> 4;

    // Scatter x tile (f32->f16) straight into A-register layout, pad c>=36.
    {
        _Float16* dst = (_Float16*)&sxA[wave][0][0];
        for (int idx = lane; idx < 16 * 64; idx += 32) {
            int r = idx >> 6, c = idx & 63;
            int row = t0 + r;
            float v = (row < Tn && c < 36) ? obs[row * 36 + c] : 0.0f;
            int kt = c >> 5, k = c & 31;
            dst[(kt * 32 + a_lane(r, k)) * 16 + a_elem(k)] = (_Float16)v;
        }
    }
    __syncthreads();

    v16h ax0 = sxA[wave][0][lane];
    v16h ax1 = sxA[wave][1][lane];

    // Stage 1: hs = relu(x @ W1 + b1); scatter D into A-register layout in LDS.
    {
        _Float16* dst = (_Float16*)&shA[wave][0][0];
#pragma unroll
        for (int nt = 0; nt < 4; ++nt) {
            float bias = b1[nt * 16 + n];
            v8f c;
#pragma unroll
            for (int r = 0; r < 8; ++r) c[r] = bias;
            c = WMMA_F32_F16(ax0, W1p[(0 * 4 + nt) * 32 + lane], c);
            c = WMMA_F32_F16(ax1, W1p[(1 * 4 + nt) * 32 + lane], c);
#pragma unroll
            for (int r = 0; r < 8; ++r) {
                float v = c[r];
                v = v > 0.0f ? v : 0.0f;               // relu
                int row = hh * 8 + r;
                int col = nt * 16 + n;
                int kt = col >> 5, k = col & 31;
                dst[(kt * 32 + a_lane(row, k)) * 16 + a_elem(k)] = (_Float16)v;
            }
        }
    }
    __syncthreads();

    v16h ah0 = shA[wave][0][lane];
    v16h ah1 = shA[wave][1][lane];

    const v16h*  Wg[3] = {Wirp, Wizp, Winp};
    const float* bg[3] = {bir, biz, bin};
    float*       og[3] = {xr, xz, xn};
#pragma unroll
    for (int g = 0; g < 3; ++g) {
#pragma unroll
        for (int nt = 0; nt < 4; ++nt) {
            float bias = bg[g][nt * 16 + n];
            v8f c;
#pragma unroll
            for (int r = 0; r < 8; ++r) c[r] = bias;
            c = WMMA_F32_F16(ah0, Wg[g][(0 * 4 + nt) * 32 + lane], c);
            c = WMMA_F32_F16(ah1, Wg[g][(1 * 4 + nt) * 32 + lane], c);
#pragma unroll
            for (int r = 0; r < 8; ++r) {
                int row = t0 + hh * 8 + r;
                if (row < Tn) og[g][row * 64 + nt * 16 + n] = c[r];
            }
        }
    }
}

// ---------------------------------------------------------------------------
// Kernel 2: sequential GRU scan. One block, 64 threads (2 waves); lane j owns
// hidden column j, its three weight columns live in VGPRs (192 regs/lane,
// only 2 waves exist so occupancy is irrelevant). h double-buffered in LDS,
// read as broadcast float4 (ds_load_b128). Gate inputs prefetched.
// h is written out already packed into the WMMA A-tile global layout.
// ---------------------------------------------------------------------------
__global__ __launch_bounds__(64) void k_scan(
        const float* __restrict__ Whr, const float* __restrict__ Whz,
        const float* __restrict__ Whn, const float* __restrict__ bhn,
        const float* __restrict__ xr, const float* __restrict__ xz,
        const float* __restrict__ xn,
        _Float16* __restrict__ hpk, int Tn) {
    __shared__ float hb[2][64];
    int j = threadIdx.x;

    float wr[64], wz[64], wn[64];
#pragma unroll
    for (int k = 0; k < 64; ++k) {
        wr[k] = Whr[k * 64 + j];
        wz[k] = Whz[k * 64 + j];
        wn[k] = Whn[k * 64 + j];
    }
    float bnj = bhn[j];
    hb[0][j]  = 0.0f;
    __syncthreads();

    // Precompute this lane's scatter position inside a packed A tile.
    int kt      = j >> 5;
    int k       = j & 31;
    int laneSel = (k >> 3) & 1;
    int elem    = a_elem(k);

    int   cur = 0;
    float gxr = xr[j], gxz = xz[j], gxn = xn[j];
    for (int t = 0; t < Tn; ++t) {
        // Prefetch next step's gate inputs; warm L2 a few steps further out.
        float nxr = 0.0f, nxz = 0.0f, nxn = 0.0f;
        if (t + 1 < Tn) {
            int o = (t + 1) * 64 + j;
            nxr = xr[o]; nxz = xz[o]; nxn = xn[o];
        }
        int tp = (t + 8 < Tn) ? (t + 8) : t;
        __builtin_prefetch(&xr[tp * 64 + j], 0, 1);
        __builtin_prefetch(&xz[tp * 64 + j], 0, 1);
        __builtin_prefetch(&xn[tp * 64 + j], 0, 1);

        float ar = 0.0f, az = 0.0f, an = 0.0f;
#pragma unroll
        for (int kq = 0; kq < 16; ++kq) {
            float4 h4 = *(const float4*)&hb[cur][kq * 4];   // broadcast b128
            ar = fmaf(h4.x, wr[kq * 4 + 0], ar);
            az = fmaf(h4.x, wz[kq * 4 + 0], az);
            an = fmaf(h4.x, wn[kq * 4 + 0], an);
            ar = fmaf(h4.y, wr[kq * 4 + 1], ar);
            az = fmaf(h4.y, wz[kq * 4 + 1], az);
            an = fmaf(h4.y, wn[kq * 4 + 1], an);
            ar = fmaf(h4.z, wr[kq * 4 + 2], ar);
            az = fmaf(h4.z, wz[kq * 4 + 2], az);
            an = fmaf(h4.z, wn[kq * 4 + 2], an);
            ar = fmaf(h4.w, wr[kq * 4 + 3], ar);
            az = fmaf(h4.w, wz[kq * 4 + 3], az);
            an = fmaf(h4.w, wn[kq * 4 + 3], an);
        }
        float r = 1.0f / (1.0f + __expf(-(gxr + ar)));
        float z = 1.0f / (1.0f + __expf(-(gxz + az)));
        float pre = gxn + r * (an + bnj);
        float e2  = __expf(-2.0f * pre);
        float nn  = (1.0f - e2) / (1.0f + e2);     // tanh
        float hj  = hb[cur][j];
        float hnew = (1.0f - z) * nn + z * hj;
        hb[cur ^ 1][j] = hnew;
        // Scatter into packed A-tile layout: tile ti = t>>4, row m = t&15.
        int m  = t & 15;
        int ti = t >> 4;
        hpk[(((ti * 2 + kt) * 32) + (m + 16 * laneSel)) * 16 + elem] = (_Float16)hnew;
        gxr = nxr; gxz = nxz; gxn = nxn;
        __syncthreads();
        cur ^= 1;
    }
}

// ---------------------------------------------------------------------------
// Kernel 3: zs = hs @ Wend' + bend' (Wend pre-permuted & padded, tiles packed).
// A tiles arrive pre-packed from the scan: two 32B vector loads per lane.
// ---------------------------------------------------------------------------
__global__ __launch_bounds__(256) void k_end(
        const v16h* __restrict__ hpk,              // packed A tiles, 2 per 16 rows
        const v16h* __restrict__ Wendp,            // 2 packed B tiles
        const float* __restrict__ bendp,           // [16] (permuted, padded)
        float* __restrict__ out, int Tn) {
    int lane = threadIdx.x & 31;
    int wave = threadIdx.x >> 5;
    int t0   = (blockIdx.x * 8 + wave) * 16;
    if (t0 >= Tn) return;                          // no __syncthreads below
    int n  = lane & 15;
    int hh = lane >> 4;
    int ti = t0 >> 4;

    v16h a0 = hpk[(ti * 2 + 0) * 32 + lane];
    v16h a1 = hpk[(ti * 2 + 1) * 32 + lane];
    v16h b0 = Wendp[0 * 32 + lane];
    v16h b1 = Wendp[1 * 32 + lane];

    float bias = bendp[n];
    v8f c;
#pragma unroll
    for (int r = 0; r < 8; ++r) c[r] = bias;
    c = WMMA_F32_F16(a0, b0, c);
    c = WMMA_F32_F16(a1, b1, c);
#pragma unroll
    for (int r = 0; r < 8; ++r) {
        int row = t0 + hh * 8 + r;
        if (row < Tn && n < 4) out[row * 4 + n] = c[r];
    }
}

// ---------------------------------------------------------------------------
// Host launcher
// ---------------------------------------------------------------------------
extern "C" void kernel_launch(void* const* d_in, const int* in_sizes, int n_in,
                              void* d_out, int out_size, void* d_ws, size_t ws_size,
                              hipStream_t stream) {
    const float* obs  = (const float*)d_in[0];
    const float* W1   = (const float*)d_in[1];
    const float* b1   = (const float*)d_in[2];
    const float* Wir  = (const float*)d_in[3];
    const float* bir  = (const float*)d_in[4];
    const float* Wiz  = (const float*)d_in[5];
    const float* biz  = (const float*)d_in[6];
    const float* Win  = (const float*)d_in[7];
    const float* bin_ = (const float*)d_in[8];
    const float* Whr  = (const float*)d_in[9];
    const float* Whz  = (const float*)d_in[10];
    const float* Whn  = (const float*)d_in[11];
    const float* bhn  = (const float*)d_in[12];
    const float* Wend = (const float*)d_in[13];
    const float* bend = (const float*)d_in[14];
    float* out = (float*)d_out;

    const int Tn = in_sizes[0] / 36;

    // Workspace layout (16B-aligned offsets)
    char* ws = (char*)d_ws;
    _Float16* W1p   = (_Float16*)(ws + 0);                 // 8 KB packed
    _Float16* Wirp  = (_Float16*)(ws + 8 * 1024);          // 8 KB packed
    _Float16* Wizp  = (_Float16*)(ws + 16 * 1024);         // 8 KB packed
    _Float16* Winp  = (_Float16*)(ws + 24 * 1024);         // 8 KB packed
    _Float16* Wendp = (_Float16*)(ws + 32 * 1024);         // 2 KB packed
    float*    bendp = (float*)   (ws + 34 * 1024);         // 64 B
    size_t gsz = (size_t)Tn * 64 * sizeof(float);
    char* p = ws + 64 * 1024;
    float* xr = (float*)p;          p += gsz;
    float* xz = (float*)p;          p += gsz;
    float* xn = (float*)p;          p += gsz;
    _Float16* hpk = (_Float16*)p;   // packed h tiles: ceil(Tn/16)*2KB == Tn*128 B

    int tiles_blocks = (Tn + 127) / 128;   // 8 waves/block, 16 rows/wave

    k_prep<<<4, 256, 0, stream>>>(W1, Wir, Wiz, Win, Wend, bend,
                                  W1p, Wirp, Wizp, Winp, Wendp, bendp);
    k_gates<<<tiles_blocks, 256, 0, stream>>>(obs, b1, bir, biz, bin_,
                                              (const v16h*)W1p, (const v16h*)Wirp,
                                              (const v16h*)Wizp, (const v16h*)Winp,
                                              xr, xz, xn, Tn);
    k_scan<<<1, 64, 0, stream>>>(Whr, Whz, Whn, bhn, xr, xz, xn, hpk, Tn);
    k_end<<<tiles_blocks, 256, 0, stream>>>((const v16h*)hpk, (const v16h*)Wendp,
                                            bendp, out, Tn);
}
